// AttentionBlock_7387343749294
// MI455X (gfx1250) — compile-verified, matrix operand order checked by reading, not run on hardware
//
#include <hip/hip_runtime.h>

// ---------------------------------------------------------------------------
// Types / WMMA helpers (CDNA5 gfx1250, wave32, 16x16x32 f16 WMMA, f32 accum)
// ---------------------------------------------------------------------------
typedef _Float16 half_t;
typedef __attribute__((ext_vector_type(16))) _Float16 v16h;
typedef __attribute__((ext_vector_type(8)))  _Float16 v8h;
typedef __attribute__((ext_vector_type(8)))  float    v8f;

#define WMMA_F16(a, b, c) \
  __builtin_amdgcn_wmma_f32_16x16x32_f16(false, (a), false, (b), (short)0, (c), false, false)

// Per-lane K index inside a 16-bit 16x32 A (or 32x16 B) fragment:
// element j (0..15) of lane L holds k = (L/16)*8 + (j&7) + (j>>3)*16
__device__ __forceinline__ int kmap_of(int lane, int j) {
  return ((lane >> 4) << 3) + (j & 7) + ((j >> 3) << 4);
}

// ---------------------------------------------------------------------------
// 1) GroupNorm: x(8,512,32,32) f32 -> h f16, 32 groups of 16 channels
// ---------------------------------------------------------------------------
__global__ __launch_bounds__(256)
void groupnorm_kernel(const float* __restrict__ x,
                      const float* __restrict__ gw,
                      const float* __restrict__ gb,
                      half_t* __restrict__ h16) {
  __shared__ float s_sum[256], s_ss[256];
  int b = blockIdx.x >> 5;
  int g = blockIdx.x & 31;
  const int GE = 16 * 1024;                 // 16 channels x 1024 spatial
  const float* xb = x + ((long long)b * 512 + g * 16) * 1024;
  float sum = 0.f, ss = 0.f;
  for (int i = threadIdx.x; i < GE; i += 256) {
    float v = xb[i];
    sum += v; ss += v * v;
  }
  s_sum[threadIdx.x] = sum; s_ss[threadIdx.x] = ss;
  __syncthreads();
  for (int off = 128; off > 0; off >>= 1) {
    if (threadIdx.x < off) {
      s_sum[threadIdx.x] += s_sum[threadIdx.x + off];
      s_ss[threadIdx.x]  += s_ss[threadIdx.x + off];
    }
    __syncthreads();
  }
  float mean = s_sum[0] * (1.0f / GE);
  float var  = s_ss[0] * (1.0f / GE) - mean * mean;
  float inv  = rsqrtf(var + 1e-5f);
  half_t* hb = h16 + ((long long)b * 512 + g * 16) * 1024;
  for (int i = threadIdx.x; i < GE; i += 256) {
    int c = g * 16 + (i >> 10);
    hb[i] = (half_t)((xb[i] - mean) * inv * gw[c] + gb[c]);
  }
}

// ---------------------------------------------------------------------------
// 2) Repack kernels: pre-swizzle plain row-major matrices into WMMA fragment
//    order so hot loops load 32 contiguous bytes per lane.
//    B-frag:  frag[((kt*NT+nt)*32+lane)*16+j] = src[(kt*32+kmap)*ncols + nt*16 + lane%16]
//      (also serves as A operand of src^T since A/B per-lane formulas mirror)
//    A-frag:  frag[((mt*KT+kt)*32+lane)*16+j] = src[(mt*16+lane%16)*kcols + kt*32 + kmap]
//    y dim batches matrices: src += (y/nInner)*outerStride + (y%nInner)*innerStride
// ---------------------------------------------------------------------------
template <typename SrcT>
__global__ __launch_bounds__(256)
void repackB_kernel(const SrcT* __restrict__ src, half_t* __restrict__ dst,
                    int KT, int NT, int ncols, int nInner,
                    long long outerStride, long long innerStride, long long dstStride) {
  int y = blockIdx.y;
  const SrcT* s = src + (long long)(y / nInner) * outerStride
                      + (long long)(y % nInner) * innerStride;
  half_t* d = dst + (long long)y * dstStride;
  int total = KT * NT * 512;
  for (int tid = blockIdx.x * 256 + threadIdx.x; tid < total; tid += gridDim.x * 256) {
    int j = tid & 15, lane = (tid >> 4) & 31, tile = tid >> 9;
    int kt = tile / NT, nt = tile - kt * NT;
    int row = kt * 32 + kmap_of(lane, j);
    int col = nt * 16 + (lane & 15);
    d[tid] = (half_t)s[(long long)row * ncols + col];
  }
}

template <typename SrcT>
__global__ __launch_bounds__(256)
void repackA_kernel(const SrcT* __restrict__ src, half_t* __restrict__ dst,
                    int MT, int KT, int kcols, int nInner,
                    long long outerStride, long long innerStride, long long dstStride) {
  int y = blockIdx.y;
  const SrcT* s = src + (long long)(y / nInner) * outerStride
                      + (long long)(y % nInner) * innerStride;
  half_t* d = dst + (long long)y * dstStride;
  int total = MT * KT * 512;
  for (int tid = blockIdx.x * 256 + threadIdx.x; tid < total; tid += gridDim.x * 256) {
    int j = tid & 15, lane = (tid >> 4) & 31, tile = tid >> 9;
    int mt = tile / KT, kt = tile - mt * KT;
    int row = mt * 16 + (lane & 15);
    int col = kt * 32 + kmap_of(lane, j);
    d[tid] = (half_t)s[(long long)row * kcols + col];
  }
}

// ---------------------------------------------------------------------------
// 3) Wave-level GEMM mainloop: one wave computes a 64(M) x 32(N) tile.
//    Afrag tiles: [mt][kt], Bfrag tiles: [kt][nt]; K step = 32.
// ---------------------------------------------------------------------------
__device__ __forceinline__
void gemm_wave64x32(const half_t* __restrict__ Afrag, const half_t* __restrict__ Bfrag,
                    int KT, int NT, int mtBase, int ntBase, int lane, v8f c[4][2]) {
  for (int kt = 0; kt < KT; ++kt) {
    v16h a[4], bb[2];
#pragma unroll
    for (int i = 0; i < 4; ++i)
      a[i] = *(const v16h*)(Afrag + (((long long)(mtBase + i) * KT + kt) * 32 + lane) * 16);
#pragma unroll
    for (int j = 0; j < 2; ++j)
      bb[j] = *(const v16h*)(Bfrag + (((long long)kt * NT + (ntBase + j)) * 32 + lane) * 16);
#pragma unroll
    for (int i = 0; i < 4; ++i)
#pragma unroll
      for (int j = 0; j < 2; ++j)
        c[i][j] = WMMA_F16(a[i], bb[j], c[i][j]);
  }
}

// QKV: out(b,1536,1024) f16 = Wq(1536x512) * h(b,512,1024) + bias
__global__ __launch_bounds__(256)
void gemm_qkv_kernel(const half_t* __restrict__ Wfrag, const half_t* __restrict__ Hfrag,
                     const float* __restrict__ bias, half_t* __restrict__ out) {
  int wave = threadIdx.x >> 5, lane = threadIdx.x & 31;
  int b = blockIdx.z;
  int wm = wave & 1, wn = wave >> 1;                // 2 x 4 waves -> 128x128 block tile
  int mBase = blockIdx.x * 128 + wm * 64;
  int nBase = blockIdx.y * 128 + wn * 32;
  v8f c[4][2];
#pragma unroll
  for (int i = 0; i < 4; ++i)
#pragma unroll
    for (int j = 0; j < 2; ++j) c[i][j] = 0.0f;
  gemm_wave64x32(Wfrag, Hfrag + (long long)b * (16LL * 64 * 512),
                 16, 64, mBase >> 4, nBase >> 4, lane, c);
  half_t* ob = out + (long long)b * 1536 * 1024;
  int rAdd = 8 * (lane >> 4), col = lane & 15;
#pragma unroll
  for (int i = 0; i < 4; ++i)
#pragma unroll
    for (int j = 0; j < 2; ++j)
#pragma unroll
      for (int r = 0; r < 8; ++r) {
        int m = mBase + i * 16 + r + rAdd;
        int n = nBase + j * 16 + col;
        ob[(long long)m * 1024 + n] = (half_t)(c[i][j][r] + bias[m]);
      }
}

// Proj: out(b,512,1024) f32 = Wp(512x512) * attnout(b,512,1024) + bias + residual x
__global__ __launch_bounds__(256)
void gemm_proj_kernel(const half_t* __restrict__ Wfrag, const half_t* __restrict__ Bfrag,
                      const float* __restrict__ bias, const float* __restrict__ resid,
                      float* __restrict__ out) {
  int wave = threadIdx.x >> 5, lane = threadIdx.x & 31;
  int b = blockIdx.z;
  int wm = wave & 1, wn = wave >> 1;
  int mBase = blockIdx.x * 128 + wm * 64;
  int nBase = blockIdx.y * 128 + wn * 32;
  v8f c[4][2];
#pragma unroll
  for (int i = 0; i < 4; ++i)
#pragma unroll
    for (int j = 0; j < 2; ++j) c[i][j] = 0.0f;
  gemm_wave64x32(Wfrag, Bfrag + (long long)b * (16LL * 64 * 512),
                 16, 64, mBase >> 4, nBase >> 4, lane, c);
  const float* xb = resid + (long long)b * 512 * 1024;
  float* ob = out + (long long)b * 512 * 1024;
  int rAdd = 8 * (lane >> 4), col = lane & 15;
#pragma unroll
  for (int i = 0; i < 4; ++i)
#pragma unroll
    for (int j = 0; j < 2; ++j)
#pragma unroll
      for (int r = 0; r < 8; ++r) {
        int m = mBase + i * 16 + r + rAdd;
        int n = nBase + j * 16 + col;
        long long idx = (long long)m * 1024 + n;
        ob[idx] = c[i][j][r] + bias[m] + xb[idx];
      }
}

// ---------------------------------------------------------------------------
// 4) Flash attention: one wave per (b, head, 16-query tile). d = 64, L = 1024.
//    Qfrag/Kfrag tiles per (b,h): [dt:2][t:64] ; Vfrag tiles: [dt:4][kt:32].
//    Per iter: 32 keys; S = Qt^T K (4 WMMA, K=64 chained), online softmax in
//    per-wave LDS, O += V * P^T (4 WMMA). Output f16 plain [b][c=512][1024].
// ---------------------------------------------------------------------------
__global__ __launch_bounds__(256)
void attention_kernel(const half_t* __restrict__ Qfrag, const half_t* __restrict__ Kfrag,
                      const half_t* __restrict__ Vfrag, half_t* __restrict__ Oplain) {
  __shared__ __align__(16) float  Stile[8][16][32];
  __shared__ __align__(16) half_t Ptile[8][16][32];
  __shared__ float mrow[8][16], lrow[8][16], arow[8][16];

  int wave = threadIdx.x >> 5, lane = threadIdx.x & 31;
  int wg = blockIdx.x * 8 + wave;              // 0..4095
  int bh = wg >> 6;                            // b*8 + head
  int qt = wg & 63;                            // query tile

  const long long qkStride = 2LL * 64 * 512;
  const long long vStride  = 4LL * 32 * 512;
  const half_t* Qb = Qfrag + (long long)bh * qkStride;
  const half_t* Kb = Kfrag + (long long)bh * qkStride;
  const half_t* Vb = Vfrag + (long long)bh * vStride;

  // Q operand (A: 16 queries x 64 d), fixed across key loop
  v16h aq0 = *(const v16h*)(Qb + ((0LL * 64 + qt) * 32 + lane) * 16);
  v16h aq1 = *(const v16h*)(Qb + ((1LL * 64 + qt) * 32 + lane) * 16);

  v8f o[4];
#pragma unroll
  for (int dt = 0; dt < 4; ++dt) o[dt] = 0.0f;

  if (lane < 16) { mrow[wave][lane] = -3.0e38f; lrow[wave][lane] = 0.f; }
  const float scale = 0.125f;                  // 64^-0.5
  int rb = (lane >> 4) * 8, col = lane & 15;

  for (int it = 0; it < 32; ++it) {
    int kb0 = it * 2, kb1 = it * 2 + 1;
    v16h bk00 = *(const v16h*)(Kb + ((0LL * 64 + kb0) * 32 + lane) * 16);
    v16h bk10 = *(const v16h*)(Kb + ((1LL * 64 + kb0) * 32 + lane) * 16);
    v16h bk01 = *(const v16h*)(Kb + ((0LL * 64 + kb1) * 32 + lane) * 16);
    v16h bk11 = *(const v16h*)(Kb + ((1LL * 64 + kb1) * 32 + lane) * 16);

    v8f sA = 0.0f, sB = 0.0f;
    sA = WMMA_F16(aq0, bk00, sA);
    sA = WMMA_F16(aq1, bk10, sA);
    sB = WMMA_F16(aq0, bk01, sB);
    sB = WMMA_F16(aq1, bk11, sB);

#pragma unroll
    for (int r = 0; r < 8; ++r) {
      Stile[wave][rb + r][col]      = sA[r] * scale;
      Stile[wave][rb + r][col + 16] = sB[r] * scale;
    }
    __syncthreads();   // (A) S visible to softmax lanes

    if (lane < 16) {
      int r = lane;
      float m0 = mrow[wave][r];
      float mx = m0;
      for (int cc = 0; cc < 32; ++cc) mx = fmaxf(mx, Stile[wave][r][cc]);
      float alpha = __expf(m0 - mx);
      float sum = 0.f;
      for (int cc = 0; cc < 32; ++cc) {
        float p = __expf(Stile[wave][r][cc] - mx);
        Ptile[wave][r][cc] = (half_t)p;
        sum += p;
      }
      lrow[wave][r] = lrow[wave][r] * alpha + sum;
      mrow[wave][r] = mx;
      arow[wave][r] = alpha;
    }
    __syncthreads();   // (B) P and alpha visible to all lanes

    float alpha = arow[wave][col];
#pragma unroll
    for (int dt = 0; dt < 4; ++dt)
#pragma unroll
      for (int r = 0; r < 8; ++r) o[dt][r] *= alpha;

    // P as B operand: B[key][q], 32 keys x 16 queries, from LDS row of P[q][key]
    union { v16h v; v8h h[2]; } bp;
    const half_t* prow = &Ptile[wave][col][0];
    bp.h[0] = *(const v8h*)(prow + ((lane >> 4) << 3));
    bp.h[1] = *(const v8h*)(prow + ((lane >> 4) << 3) + 16);

#pragma unroll
    for (int dt = 0; dt < 4; ++dt) {
      v16h av = *(const v16h*)(Vb + (((long long)dt * 32 + it) * 32 + lane) * 16);
      o[dt] = WMMA_F16(av, bp.v, o[dt]);
    }
  }

  float inv = 1.0f / lrow[wave][col];
  int b = bh >> 3, head = bh & 7;
  half_t* Ob = Oplain + ((long long)b * 512 + head * 64) * 1024;
#pragma unroll
  for (int dt = 0; dt < 4; ++dt)
#pragma unroll
    for (int r = 0; r < 8; ++r) {
      int d_ = dt * 16 + r + rb;
      int n  = qt * 16 + col;
      Ob[(long long)d_ * 1024 + n] = (half_t)(o[dt][r] * inv);
    }
}

// ---------------------------------------------------------------------------
// Launch
// ---------------------------------------------------------------------------
extern "C" void kernel_launch(void* const* d_in, const int* in_sizes, int n_in,
                              void* d_out, int out_size, void* d_ws, size_t ws_size,
                              hipStream_t stream) {
  (void)in_sizes; (void)n_in; (void)out_size; (void)ws_size;
  const float* x      = (const float*)d_in[0];
  const float* gn_w   = (const float*)d_in[1];
  const float* gn_b   = (const float*)d_in[2];
  const float* qkv_w  = (const float*)d_in[3];
  const float* qkv_b  = (const float*)d_in[4];
  const float* proj_w = (const float*)d_in[5];
  const float* proj_b = (const float*)d_in[6];
  float* out = (float*)d_out;

  half_t* ws = (half_t*)d_ws;
  long long off = 0;
  half_t* h16   = ws + off; off += 8LL * 512 * 1024;     // groupnorm out (plain)
  half_t* Hfrag = ws + off; off += 8LL * 512 * 1024;     // h B-frags
  half_t* Wq    = ws + off; off += 1536LL * 512;         // qkv_w A-frags
  half_t* Wp    = ws + off; off += 512LL * 512;          // proj_w A-frags
  half_t* qkvp  = ws + off; off += 8LL * 1536 * 1024;    // qkv out (plain)
  half_t* Qf    = ws + off; off += 64LL * 2 * 64 * 512;  // Q frags per (b,h)
  half_t* Kf    = ws + off; off += 64LL * 2 * 64 * 512;  // K frags per (b,h)
  half_t* Vf    = ws + off; off += 64LL * 4 * 32 * 512;  // V frags per (b,h)
  half_t* Op    = ws + off; off += 8LL * 512 * 1024;     // attention out (plain)
  half_t* Of    = ws + off; off += 8LL * 512 * 1024;     // attention out B-frags

  // 1) GroupNorm
  groupnorm_kernel<<<dim3(256), dim3(256), 0, stream>>>(x, gn_w, gn_b, h16);

  // 2) Weight repacks (f32 -> f16 A-frags)
  repackA_kernel<float><<<dim3(768, 1), dim3(256), 0, stream>>>(
      qkv_w, Wq, 96, 16, 512, 1, 0, 0, 0);
  repackA_kernel<float><<<dim3(256, 1), dim3(256), 0, stream>>>(
      proj_w, Wp, 32, 16, 512, 1, 0, 0, 0);

  // 3) h -> B-frags (per batch)
  repackB_kernel<half_t><<<dim3(512, 8), dim3(256), 0, stream>>>(
      h16, Hfrag, 16, 64, 1024, 1, 512LL * 1024, 0, 16LL * 64 * 512);

  // 4) QKV GEMM: (1536x512)x(512x1024) per batch
  gemm_qkv_kernel<<<dim3(12, 8, 8), dim3(256), 0, stream>>>(Wq, Hfrag, qkv_b, qkvp);

  // 5) Q/K/V repacks per (b, head): 64 matrices, inner stride = head (64 rows)
  repackB_kernel<half_t><<<dim3(64, 64), dim3(256), 0, stream>>>(
      qkvp, Qf, 2, 64, 1024, 8, 1536LL * 1024, 64LL * 1024, 2LL * 64 * 512);
  repackB_kernel<half_t><<<dim3(64, 64), dim3(256), 0, stream>>>(
      qkvp + 512LL * 1024, Kf, 2, 64, 1024, 8, 1536LL * 1024, 64LL * 1024, 2LL * 64 * 512);
  repackA_kernel<half_t><<<dim3(64, 64), dim3(256), 0, stream>>>(
      qkvp + 1024LL * 1024, Vf, 4, 32, 1024, 8, 1536LL * 1024, 64LL * 1024, 4LL * 32 * 512);

  // 6) Flash attention: 4096 waves = 512 blocks x 8 waves
  attention_kernel<<<dim3(512), dim3(256), 0, stream>>>(Qf, Kf, Vf, Op);

  // 7) Attention out -> B-frags, then proj GEMM + bias + residual -> f32 out
  repackB_kernel<half_t><<<dim3(512, 8), dim3(256), 0, stream>>>(
      Op, Of, 16, 64, 1024, 1, 512LL * 1024, 0, 16LL * 64 * 512);
  gemm_proj_kernel<<<dim3(4, 8, 8), dim3(256), 0, stream>>>(Wp, Of, proj_b, x, out);
}
